// Mamba2Layer_27685359190812
// MI455X (gfx1250) — compile-verified, matrix operand order checked by reading, not run on hardware
//
#include <hip/hip_runtime.h>
#include <hip/hip_bf16.h>
#include <stdint.h>

#define D_MODEL 1024
#define D_STATE 128
#define D_CONV  4
#define D_INNER 2048
#define DT_RANK 64
#define BATCH   2
#define SEQ     2048
#define NTOK    (BATCH * SEQ)            // 4096 tokens
#define XDBL_N  (DT_RANK + 2 * D_STATE)  // 320

typedef __bf16 bf16;
typedef __attribute__((ext_vector_type(16))) __bf16 bf16x16;
typedef __attribute__((ext_vector_type(8)))  float  f32x8;

union FragU { bf16x16 v; uint4 q[2]; };

__device__ __forceinline__ bf16 f2bf(float f) {
  unsigned u = __builtin_bit_cast(unsigned, f);
  unsigned r = (u + 0x7FFFu + ((u >> 16) & 1u)) >> 16;  // round-to-nearest-even
  unsigned short s = (unsigned short)r;
  return __builtin_bit_cast(bf16, s);
}

// ---------------------------------------------------------------------------
// C[M,N] = A[M,K](bf16,row-major) * B[N,K](bf16,row-major)^T, fp32 accum.
// Block = 8 waves (256 thr), block tile 128(M)x64(N).
// Wave tile 32x32: 2 A frags x 2 B frags -> 4 v_wmma per K-step
// (~16 FLOP/byte of fragment traffic, 2x round-1).
// All M,N,K in this layer are multiples of 128/64/32 -> no bounds checks.
// Fragment layout per CDNA5 ISA 7.12.2 (16-bit A/B 16x32):
//   lanes 0-15  : elems 0-7 = K[0..7],  elems 8-15 = K[16..23]
//   lanes 16-31 : elems 0-7 = K[8..15], elems 8-15 = K[24..31]
// ---------------------------------------------------------------------------
__global__ __launch_bounds__(256, 2)
void wmma_gemm_bf16_nt(const bf16* __restrict__ A, const bf16* __restrict__ B,
                       float* __restrict__ C, int M, int N, int K) {
  const int lane = threadIdx.x & 31;
  const int wave = threadIdx.x >> 5;
  const int wm = wave & 3;   // 4 waves along M, 32 rows each
  const int wn = wave >> 2;  // 2 waves along N, 32 cols each
  const int tileM0 = blockIdx.y * 128 + wm * 32;
  const int tileN0 = blockIdx.x * 64  + wn * 32;
  const int r  = lane & 15;
  const int hi = lane >> 4;
  const int kh = hi * 8;

  const bf16* pa0 = A + (long)(tileM0 + r)      * K + kh;
  const bf16* pa1 = A + (long)(tileM0 + 16 + r) * K + kh;
  const bf16* pb0 = B + (long)(tileN0 + r)      * K + kh;
  const bf16* pb1 = B + (long)(tileN0 + 16 + r) * K + kh;

  f32x8 acc00 = {0.f,0.f,0.f,0.f,0.f,0.f,0.f,0.f};
  f32x8 acc01 = acc00, acc10 = acc00, acc11 = acc00;

  for (int k0 = 0; k0 < K; k0 += 32) {
    FragU fa0, fa1, fb0, fb1;
    fa0.q[0] = *(const uint4*)(pa0);        // K[kh..kh+7]
    fa0.q[1] = *(const uint4*)(pa0 + 16);   // K[16+kh..16+kh+7]
    fa1.q[0] = *(const uint4*)(pa1);
    fa1.q[1] = *(const uint4*)(pa1 + 16);
    fb0.q[0] = *(const uint4*)(pb0);
    fb0.q[1] = *(const uint4*)(pb0 + 16);
    fb1.q[0] = *(const uint4*)(pb1);
    fb1.q[1] = *(const uint4*)(pb1 + 16);
    acc00 = __builtin_amdgcn_wmma_f32_16x16x32_bf16(false, fa0.v, false, fb0.v,
                                                    (short)0, acc00, false, false);
    acc01 = __builtin_amdgcn_wmma_f32_16x16x32_bf16(false, fa0.v, false, fb1.v,
                                                    (short)0, acc01, false, false);
    acc10 = __builtin_amdgcn_wmma_f32_16x16x32_bf16(false, fa1.v, false, fb0.v,
                                                    (short)0, acc10, false, false);
    acc11 = __builtin_amdgcn_wmma_f32_16x16x32_bf16(false, fa1.v, false, fb1.v,
                                                    (short)0, acc11, false, false);
    // Prefetch streamed activations ~8 K-steps ahead (weights are L2-hot).
    __builtin_prefetch(pa0 + 256, 0, 1);   // global_prefetch_b8
    __builtin_prefetch(pa1 + 256, 0, 1);
    pa0 += 32; pa1 += 32; pb0 += 32; pb1 += 32;
  }

  // C/D layout: VGPR i holds row (i + 8*hi), col = lane&15
  float* c00 = C + (long)(tileM0 + hi * 8)      * N + tileN0 + r;
  float* c10 = C + (long)(tileM0 + 16 + hi * 8) * N + tileN0 + r;
#pragma unroll
  for (int i = 0; i < 8; ++i) {
    c00[(long)i * N]      = acc00[i];
    c00[(long)i * N + 16] = acc01[i];
    c10[(long)i * N]      = acc10[i];
    c10[(long)i * N + 16] = acc11[i];
  }
}

// --------------------------- elementwise helpers ---------------------------
__global__ void f32_to_bf16_k(const float* __restrict__ s, bf16* __restrict__ d, long n) {
  long i = (long)blockIdx.x * blockDim.x + threadIdx.x;
  if (i < n) d[i] = f2bf(s[i]);
}

__global__ void neg_exp_k(const float* __restrict__ s, float* __restrict__ d, long n) {
  long i = (long)blockIdx.x * blockDim.x + threadIdx.x;
  if (i < n) d[i] = -__expf(s[i]);
}

// u = silu(depthwise_causal_conv(xc) + bias); xc = xz[..., :D_INNER]
__global__ __launch_bounds__(256)
void conv_silu_k(const float* __restrict__ xz, const float* __restrict__ w,
                 const float* __restrict__ bias, float* __restrict__ u,
                 bf16* __restrict__ ubf) {
  long idx = (long)blockIdx.x * blockDim.x + threadIdx.x;
  if (idx >= (long)NTOK * D_INNER) return;
  const int  d   = (int)(idx & (D_INNER - 1));
  const long tok = idx >> 11;          // D_INNER = 2^11
  const int  t   = (int)(tok & (SEQ - 1));
  const int  b   = (int)(tok >> 11);   // SEQ = 2^11
  float acc = bias[d];
#pragma unroll
  for (int j = 0; j < D_CONV; ++j) {
    const int tt = t - (D_CONV - 1) + j;
    if (tt >= 0)
      acc += w[d * D_CONV + j] * xz[((long)b * SEQ + tt) * (2 * D_INNER) + d];
  }
  const float s = acc / (1.f + __expf(-acc));
  u[idx]   = s;
  ubf[idx] = f2bf(s);
}

__global__ void extract_dtlo_k(const float* __restrict__ xdbl, bf16* __restrict__ dtlo) {
  long i = (long)blockIdx.x * blockDim.x + threadIdx.x;
  if (i >= (long)NTOK * DT_RANK) return;
  const long tok = i >> 6;             // DT_RANK = 64
  const int  rr  = (int)(i & 63);
  dtlo[i] = f2bf(xdbl[tok * XDBL_N + rr]);
}

__global__ void softplus_k(float* __restrict__ dt, const float* __restrict__ bdt) {
  long i = (long)blockIdx.x * blockDim.x + threadIdx.x;
  if (i >= (long)NTOK * D_INNER) return;
  const int d = (int)(i & (D_INNER - 1));
  const float v = dt[i] + 2.0f * bdt[d];
  dt[i] = (v > 20.f) ? v : log1pf(__expf(v));
}

// ---------------------------------------------------------------------------
// Fused selective scan. One wave per (batch, d_inner) channel: 4096 waves.
// Each lane keeps 4 of 128 states + 4 A coeffs in registers. B_t/C_t are
// shared by all 2048 channel-waves of a batch -> L2 broadcast. The loop is
// strictly sequential, so prefetch B/C 8 steps ahead to pull L2 latency off
// the dependence chain. Fuses y = (h.C + u*Dp) * silu(z) and the bf16
// down-convert feeding the output WMMA GEMM.
// ---------------------------------------------------------------------------
__global__ __launch_bounds__(256)
void scan_fused_k(const float* __restrict__ dt, const float* __restrict__ xdbl,
                  const float* __restrict__ u,  const float* __restrict__ A,
                  const float* __restrict__ Dp, const float* __restrict__ xz,
                  bf16* __restrict__ ybf) {
  const int lane = threadIdx.x & 31;
  const int gw   = blockIdx.x * (blockDim.x >> 5) + (threadIdx.x >> 5);
  const int b    = gw >> 11;             // D_INNER = 2^11
  const int d    = gw & (D_INNER - 1);

  const float4 a = *(const float4*)(A + (long)d * D_STATE + lane * 4);
  const float  dp = Dp[d];
  float4 h = {0.f, 0.f, 0.f, 0.f};

  const float* bc  = xdbl + (long)b * SEQ * XDBL_N + DT_RANK + lane * 4;
  const float* dtp = dt + (long)b * SEQ * D_INNER + d;
  const float* up  = u  + (long)b * SEQ * D_INNER + d;
  const float* zp  = xz + (long)b * SEQ * (2 * D_INNER) + D_INNER + d;
  bf16*        yp  = ybf + (long)b * SEQ * D_INNER + d;

  for (int t = 0; t < SEQ; ++t) {
    // Hide L2 latency of the serial chain: prefetch step t+8.
    if (t + 8 < SEQ) {
      __builtin_prefetch(bc + (long)(t + 8) * XDBL_N, 0, 1);
      __builtin_prefetch(bc + (long)(t + 8) * XDBL_N + D_STATE, 0, 1);
      __builtin_prefetch(dtp + (long)(t + 8) * D_INNER, 0, 1);
    }
    const float  dtv = dtp[(long)t * D_INNER];
    const float  uv  = up [(long)t * D_INNER];
    const float4 Bt  = *(const float4*)(bc + (long)t * XDBL_N);
    const float4 Ct  = *(const float4*)(bc + (long)t * XDBL_N + D_STATE);
    const float  du  = dtv * uv;
    h.x = __expf(dtv * a.x) * h.x + du * Bt.x;
    h.y = __expf(dtv * a.y) * h.y + du * Bt.y;
    h.z = __expf(dtv * a.z) * h.z + du * Bt.z;
    h.w = __expf(dtv * a.w) * h.w + du * Bt.w;
    float part = h.x * Ct.x + h.y * Ct.y + h.z * Ct.z + h.w * Ct.w;
#pragma unroll
    for (int off = 16; off > 0; off >>= 1)
      part += __shfl_xor(part, off, 32);
    if (lane == 0) {
      float y  = part + uv * dp;
      const float zv = zp[(long)t * (2 * D_INNER)];
      y *= zv / (1.f + __expf(-zv));
      yp[(long)t * D_INNER] = f2bf(y);
    }
  }
}

// ---------------------------------------------------------------------------
extern "C" void kernel_launch(void* const* d_in, const int* in_sizes, int n_in,
                              void* d_out, int out_size, void* d_ws, size_t ws_size,
                              hipStream_t stream) {
  (void)in_sizes; (void)n_in; (void)out_size; (void)ws_size;
  const float* x      = (const float*)d_in[0];
  const float* W_in   = (const float*)d_in[1];
  const float* W_out  = (const float*)d_in[2];
  const float* conv_w = (const float*)d_in[3];
  const float* conv_b = (const float*)d_in[4];
  const float* W_x    = (const float*)d_in[5];
  const float* W_dt   = (const float*)d_in[6];
  const float* b_dt   = (const float*)d_in[7];
  const float* A_log  = (const float*)d_in[8];
  const float* Dp     = (const float*)d_in[9];
  float* out = (float*)d_out;

  char* wsp = (char*)d_ws;
  auto alloc = [&](size_t bytes) -> char* {
    char* p = wsp; wsp += (bytes + 255) & ~(size_t)255; return p;
  };
  bf16*  x_bf    = (bf16*) alloc((size_t)NTOK * D_MODEL * 2);
  bf16*  Win_bf  = (bf16*) alloc((size_t)2 * D_INNER * D_MODEL * 2);
  bf16*  Wx_bf   = (bf16*) alloc((size_t)XDBL_N * D_INNER * 2);
  bf16*  Wdt_bf  = (bf16*) alloc((size_t)D_INNER * DT_RANK * 2);
  bf16*  Wout_bf = (bf16*) alloc((size_t)D_MODEL * D_INNER * 2);
  float* xz      = (float*)alloc((size_t)NTOK * 2 * D_INNER * 4);
  float* u_f     = (float*)alloc((size_t)NTOK * D_INNER * 4);
  bf16*  u_bf    = (bf16*) alloc((size_t)NTOK * D_INNER * 2);
  float* xdbl    = (float*)alloc((size_t)NTOK * XDBL_N * 4);
  bf16*  dtlo_bf = (bf16*) alloc((size_t)NTOK * DT_RANK * 2);
  float* dt_f    = (float*)alloc((size_t)NTOK * D_INNER * 4);
  float* A_f     = (float*)alloc((size_t)D_INNER * D_STATE * 4);
  bf16*  y_bf    = (bf16*) alloc((size_t)NTOK * D_INNER * 2);

  auto cvt = [&](const float* s, bf16* d, long n) {
    f32_to_bf16_k<<<dim3((unsigned)((n + 255) / 256)), dim3(256), 0, stream>>>(s, d, n);
  };
  cvt(x,     x_bf,    (long)NTOK * D_MODEL);
  cvt(W_in,  Win_bf,  (long)2 * D_INNER * D_MODEL);
  cvt(W_x,   Wx_bf,   (long)XDBL_N * D_INNER);
  cvt(W_dt,  Wdt_bf,  (long)D_INNER * DT_RANK);
  cvt(W_out, Wout_bf, (long)D_MODEL * D_INNER);

  {
    long n = (long)D_INNER * D_STATE;
    neg_exp_k<<<dim3((unsigned)((n + 255) / 256)), dim3(256), 0, stream>>>(A_log, A_f, n);
  }

  auto gemm = [&](const bf16* A, const bf16* B, float* C, int M, int N, int K) {
    wmma_gemm_bf16_nt<<<dim3(N / 64, M / 128), dim3(256), 0, stream>>>(A, B, C, M, N, K);
  };

  // xz = x @ W_in^T : (4096,1024) x (4096,1024)^T -> (4096,4096)
  gemm(x_bf, Win_bf, xz, NTOK, 2 * D_INNER, D_MODEL);

  // u = silu(conv(xc) + b)
  {
    long n = (long)NTOK * D_INNER;
    conv_silu_k<<<dim3((unsigned)((n + 255) / 256)), dim3(256), 0, stream>>>(
        xz, conv_w, conv_b, u_f, u_bf);
  }

  // x_dbl = u @ W_x^T : (4096,2048) x (320,2048)^T -> (4096,320)
  gemm(u_bf, Wx_bf, xdbl, NTOK, XDBL_N, D_INNER);

  {
    long n = (long)NTOK * DT_RANK;
    extract_dtlo_k<<<dim3((unsigned)((n + 255) / 256)), dim3(256), 0, stream>>>(xdbl, dtlo_bf);
  }

  // dt_lin = dt_lo @ W_dt^T : (4096,64) x (2048,64)^T -> (4096,2048)
  gemm(dtlo_bf, Wdt_bf, dt_f, NTOK, D_INNER, DT_RANK);

  {
    long n = (long)NTOK * D_INNER;
    softplus_k<<<dim3((unsigned)((n + 255) / 256)), dim3(256), 0, stream>>>(dt_f, b_dt);
  }

  // fused selective scan + gating -> y_bf
  scan_fused_k<<<dim3((BATCH * D_INNER) / 8), dim3(256), 0, stream>>>(
      dt_f, xdbl, u_f, A_f, Dp, xz, y_bf);

  // out = y @ W_out^T : (4096,2048) x (1024,2048)^T -> (4096,1024)
  gemm(y_bf, Wout_bf, out, NTOK, D_MODEL, D_INNER);
}
